// UOTLoss_12970801234487
// MI455X (gfx1250) — matrix operand order repaired and checked by reading, not compiled, not run on hardware
//
#include <hip/hip_runtime.h>
#include <hip/hip_bf16.h>

typedef __attribute__((ext_vector_type(16))) __bf16 v16bf;
typedef __attribute__((ext_vector_type(8)))  __bf16 v8bf;
typedef __attribute__((ext_vector_type(4)))  __bf16 v4bf;
typedef __attribute__((ext_vector_type(8)))  float  v8f;

#define B_   128
#define N_   576
#define M_   256
#define D_   1024
#define EPS  0.1f
#define INV_EPS 10.0f
#define LSTR 40   // padded LDS row stride in bf16 elements (80 bytes, 16B-aligned chunks)

__device__ __forceinline__ float waveReduceSum(float v) {
#pragma unroll
    for (int m = 16; m >= 1; m >>= 1) v += __shfl_xor(v, m, 32);
    return v;
}

// ---------------------------------------------------------------------------
// Kernel 1: inverse L2 norms of every row of x and y. One wave32 per row.
// ---------------------------------------------------------------------------
__global__ __launch_bounds__(256) void uot_norms_kernel(const float* __restrict__ x,
                                                        const float* __restrict__ y,
                                                        float* __restrict__ xinv,
                                                        float* __restrict__ yinv) {
    const int lane = threadIdx.x & 31;
    const int wave = threadIdx.x >> 5;
    const int row  = blockIdx.x * 8 + wave;

    const float4* p4;
    float* outp;
    if (row < B_ * N_) {
        p4   = (const float4*)(x + (size_t)row * D_);
        outp = xinv + row;
    } else {
        const int r2 = row - B_ * N_;
        p4   = (const float4*)(y + (size_t)r2 * D_);
        outp = yinv + r2;
    }

    float acc = 0.0f;
#pragma unroll
    for (int j = 0; j < 8; ++j) {
        float4 q = p4[lane + 32 * j];
        acc += q.x * q.x + q.y * q.y + q.z * q.z + q.w * q.w;
    }
    acc = waveReduceSum(acc);
    if (lane == 0) *outp = 1.0f / fmaxf(sqrtf(acc), 1e-12f);
}

// ---------------------------------------------------------------------------
// Kernel 2: cost GEMM with fused normalize->bf16 and K = exp((s-1)/eps).
// 64x64 tile per 256-thread block; 8 waves in 4(n) x 2(m); each wave does
// two v_wmma_f32_16x16x32_bf16 per 32-wide K slab staged in LDS.
// ---------------------------------------------------------------------------
__global__ __launch_bounds__(256) void uot_costK_kernel(const float* __restrict__ x,
                                                        const float* __restrict__ y,
                                                        const float* __restrict__ xinv,
                                                        const float* __restrict__ yinv,
                                                        float* __restrict__ Kmat) {
    __shared__ __bf16 As[64 * LSTR];
    __shared__ __bf16 Bs[64 * LSTR];

    const int b     = blockIdx.z;
    const int nBase = blockIdx.y * 64;
    const int mBase = blockIdx.x * 64;

    const int tid  = threadIdx.x;
    const int lane = tid & 31;
    const int wave = tid >> 5;
    const int wn   = wave >> 1;      // 0..3 : 16-row slab inside tile
    const int wm   = wave & 1;       // 0..1 : 32-col slab inside tile
    const int lrow = lane & 15;
    const int hi   = lane >> 4;      // 0 or 1 (half-wave)

    // staging map: each thread loads two float4 per matrix per K-slab
    const int ldRow = tid >> 3;          // 0..31
    const int ldCol = (tid & 7) << 2;    // 0,4,...,28

    const float* xRow0 = x + ((size_t)b * N_ + nBase + ldRow) * D_ + ldCol;
    const float* xRow1 = xRow0 + (size_t)32 * D_;
    const float* yRow0 = y + ((size_t)b * M_ + mBase + ldRow) * D_ + ldCol;
    const float* yRow1 = yRow0 + (size_t)32 * D_;
    const float sA0 = xinv[b * N_ + nBase + ldRow];
    const float sA1 = xinv[b * N_ + nBase + ldRow + 32];
    const float sB0 = yinv[b * M_ + mBase + ldRow];
    const float sB1 = yinv[b * M_ + mBase + ldRow + 32];

    v8f acc0 = {};
    v8f acc1 = {};

    for (int k0 = 0; k0 < D_; k0 += 32) {
        // ---- stage normalized bf16 tiles into LDS ----
        {
            float4 xa0 = *(const float4*)(xRow0 + k0);
            float4 xa1 = *(const float4*)(xRow1 + k0);
            float4 ya0 = *(const float4*)(yRow0 + k0);
            float4 ya1 = *(const float4*)(yRow1 + k0);
            v4bf pa0, pa1, pb0, pb1;
            pa0[0] = (__bf16)(xa0.x * sA0); pa0[1] = (__bf16)(xa0.y * sA0);
            pa0[2] = (__bf16)(xa0.z * sA0); pa0[3] = (__bf16)(xa0.w * sA0);
            pa1[0] = (__bf16)(xa1.x * sA1); pa1[1] = (__bf16)(xa1.y * sA1);
            pa1[2] = (__bf16)(xa1.z * sA1); pa1[3] = (__bf16)(xa1.w * sA1);
            pb0[0] = (__bf16)(ya0.x * sB0); pb0[1] = (__bf16)(ya0.y * sB0);
            pb0[2] = (__bf16)(ya0.z * sB0); pb0[3] = (__bf16)(ya0.w * sB0);
            pb1[0] = (__bf16)(ya1.x * sB1); pb1[1] = (__bf16)(ya1.y * sB1);
            pb1[2] = (__bf16)(ya1.z * sB1); pb1[3] = (__bf16)(ya1.w * sB1);
            *(v4bf*)&As[ldRow * LSTR + ldCol]        = pa0;
            *(v4bf*)&As[(ldRow + 32) * LSTR + ldCol] = pa1;
            *(v4bf*)&Bs[ldRow * LSTR + ldCol]        = pb0;
            *(v4bf*)&Bs[(ldRow + 32) * LSTR + ldCol] = pb1;
        }
        __syncthreads();

        // ---- build fragments per ISA 7.12.2 layouts ----
        // A 16x32 bf16: lanes 0-15 hold K{0-7,16-23}, lanes 16-31 hold K{8-15,24-31}
        const int aOff = (wn * 16 + lrow) * LSTR + (hi ? 8 : 0);
        v8bf a0 = *(const v8bf*)&As[aOff];
        v8bf a1 = *(const v8bf*)&As[aOff + 16];
        v16bf afrag;
#pragma unroll
        for (int i = 0; i < 8; ++i) { afrag[i] = a0[i]; afrag[8 + i] = a1[i]; }

        // B 32x16 bf16: lanes 0-15 hold K{0-7,16-23}, lanes 16-31 hold K{8-15,24-31}; col = lane%16
        const int kb    = hi ? 8 : 0;
        const int bOff0 = (wm * 32 + lrow) * LSTR + kb;
        const int bOff1 = (wm * 32 + 16 + lrow) * LSTR + kb;
        v8bf b00 = *(const v8bf*)&Bs[bOff0];
        v8bf b01 = *(const v8bf*)&Bs[bOff0 + 16];
        v8bf b10 = *(const v8bf*)&Bs[bOff1];
        v8bf b11 = *(const v8bf*)&Bs[bOff1 + 16];
        v16bf bf0, bf1;
#pragma unroll
        for (int i = 0; i < 8; ++i) {
            bf0[i] = b00[i]; bf0[8 + i] = b01[i];
            bf1[i] = b10[i]; bf1[8 + i] = b11[i];
        }

        acc0 = __builtin_amdgcn_wmma_f32_16x16x32_bf16(false, afrag, false, bf0,
                                                       (short)0, acc0, false, false);
        acc1 = __builtin_amdgcn_wmma_f32_16x16x32_bf16(false, afrag, false, bf1,
                                                       (short)0, acc1, false, false);
        __syncthreads();
    }

    // ---- epilogue: K = exp((s - 1) / eps); D layout: M = v + 8*hi, N = lane%16 ----
    const int nRow  = nBase + wn * 16 + hi * 8;
    const int mCol  = mBase + wm * 32 + lrow;
    float* outRow = Kmat + ((size_t)b * N_ + nRow) * M_;
#pragma unroll
    for (int v = 0; v < 8; ++v) {
        outRow[(size_t)v * M_ + mCol]      = __expf((acc0[v] - 1.0f) * INV_EPS);
        outRow[(size_t)v * M_ + mCol + 16] = __expf((acc1[v] - 1.0f) * INV_EPS);
    }
}

// ---------------------------------------------------------------------------
// Kernel 3: 50 Sinkhorn iterations + per-batch loss. One 256-thread block per
// batch; K stays L2-resident (75.5 MB total). fi = 0.5 -> sqrt.
// ---------------------------------------------------------------------------
__global__ __launch_bounds__(256) void uot_sinkhorn_kernel(const int* __restrict__ y_mask,
                                                           const float* __restrict__ Kmat,
                                                           float* __restrict__ bloss) {
    __shared__ float u_s[N_];
    __shared__ float v_s[M_];
    __shared__ float nu_s[M_];
    __shared__ float red[8];
    __shared__ float sh_scalar;

    const int b    = blockIdx.x;
    const int t    = threadIdx.x;     // 0..255 (== column index m)
    const int lane = t & 31;
    const int wave = t >> 5;

    const float* Kb = Kmat + (size_t)b * N_ * M_;

    // nu = mask / (sum(mask) + 1e-8), fixed-order reduction
    const float mval = (float)y_mask[b * M_ + t];
    float s = waveReduceSum(mval);
    if (lane == 0) red[wave] = s;
    __syncthreads();
    if (t == 0) {
        float tot = 0.0f;
#pragma unroll
        for (int i = 0; i < 8; ++i) tot += red[i];
        sh_scalar = tot;
    }
    __syncthreads();
    nu_s[t] = mval / (sh_scalar + 1e-8f);
    v_s[t]  = 0.0f;                   // reference starts with v = 0
    __syncthreads();

    const float MU = 1.0f / (float)N_;

    for (int it = 0; it < 50; ++it) {
        // registerize v for this iteration
        float vr[8];
#pragma unroll
        for (int j = 0; j < 8; ++j) vr[j] = v_s[lane + 32 * j];

        // u = sqrt(mu / (K v + 1e-8)) : wave-per-row, coalesced along m
        for (int r = wave; r < N_; r += 8) {
            const float* row = Kb + (size_t)r * M_;
            float acc = 0.0f;
#pragma unroll
            for (int j = 0; j < 8; ++j) acc += row[lane + 32 * j] * vr[j];
            acc = waveReduceSum(acc);
            if (lane == 0) u_s[r] = sqrtf(MU / (acc + 1e-8f));
        }
        __syncthreads();

        // v = sqrt(nu / (K^T u + 1e-8)) : thread-per-column, coalesced along m
        float acc = 0.0f;
        const float* col = Kb + t;
#pragma unroll 4
        for (int n = 0; n < N_; ++n) acc += col[(size_t)n * M_] * u_s[n];
        v_s[t] = sqrtf(nu_s[t] / (acc + 1e-8f));
        __syncthreads();
    }

    // loss_b = sum_{n,m} u[n] * K * v[m] * cost,  cost = -eps * ln(K)
    float lacc = 0.0f;
    {
        const float* col = Kb + t;
#pragma unroll 4
        for (int n = 0; n < N_; ++n) {
            const float kv   = col[(size_t)n * M_];
            const float cost = -EPS * __logf(fmaxf(kv, 1e-30f));
            lacc += u_s[n] * kv * cost;
        }
        lacc *= v_s[t];
    }
    lacc = waveReduceSum(lacc);
    if (lane == 0) red[wave] = lacc;
    __syncthreads();
    if (t == 0) {
        float tot = 0.0f;
#pragma unroll
        for (int i = 0; i < 8; ++i) tot += red[i];
        bloss[b] = tot;
    }
}

// ---------------------------------------------------------------------------
// Kernel 4: mean over batches (deterministic fixed-order reduce).
// ---------------------------------------------------------------------------
__global__ __launch_bounds__(128) void uot_finalize_kernel(const float* __restrict__ bloss,
                                                           float* __restrict__ out) {
    __shared__ float red[4];
    const int t = threadIdx.x;        // 128 threads = 4 waves
    float v = bloss[t];
    v = waveReduceSum(v);
    if ((t & 31) == 0) red[t >> 5] = v;
    __syncthreads();
    if (t == 0) out[0] = (red[0] + red[1] + red[2] + red[3]) * (1.0f / (float)B_);
}

// ---------------------------------------------------------------------------
extern "C" void kernel_launch(void* const* d_in, const int* in_sizes, int n_in,
                              void* d_out, int out_size, void* d_ws, size_t ws_size,
                              hipStream_t stream) {
    const float* x      = (const float*)d_in[0];
    const float* y      = (const float*)d_in[1];
    const int*   y_mask = (const int*)d_in[2];
    float*       out    = (float*)d_out;

    // workspace layout: K (75.5 MB) | xinv | yinv | bloss
    float* Kmat = (float*)d_ws;
    float* xinv = Kmat + (size_t)B_ * N_ * M_;
    float* yinv = xinv + (size_t)B_ * N_;
    float* bloss = yinv + (size_t)B_ * M_;

    // 1) inverse norms: (B*N + B*M) rows, 8 rows per 256-thread block
    const int totalRows = B_ * N_ + B_ * M_;          // 106496
    uot_norms_kernel<<<totalRows / 8, 256, 0, stream>>>(x, y, xinv, yinv);

    // 2) cost GEMM + exp -> K   (tiles: 4 in m, 9 in n, 128 batches)
    dim3 gGemm(M_ / 64, N_ / 64, B_);
    uot_costK_kernel<<<gGemm, 256, 0, stream>>>(x, y, xinv, yinv, Kmat);

    // 3) Sinkhorn (50 iters) + per-batch loss
    uot_sinkhorn_kernel<<<B_, 256, 0, stream>>>(y_mask, Kmat, bloss);

    // 4) mean over batches
    uot_finalize_kernel<<<1, 128, 0, stream>>>(bloss, out);
}
